// CrossAttention_82944408420441
// MI455X (gfx1250) — matrix operand order repaired and checked by reading, not compiled, not run on hardware
//
#include <hip/hip_runtime.h>
#include <hip/hip_bf16.h>

#define D_MODEL 1024
#define DIMS    64
#define BATCH   4
#define SQ      2048
#define SK      4096
#define MQ      (BATCH * SQ)   // 8192 query rows
#define MK      (BATCH * SK)   // 16384 key/value rows

#define NSPLIT  4              // flash split-K partitions over Sk
#define SKP     (SK / NSPLIT)  // 1024 keys per partition
#define CHUNK   64             // keys per flash inner iteration

typedef __attribute__((ext_vector_type(16))) __bf16 bf16x16;
typedef __attribute__((ext_vector_type(8)))  __bf16 bf16x8;
typedef __attribute__((ext_vector_type(8)))  float  f32x8;

static __device__ inline bf16x16 bjoin(bf16x8 lo, bf16x8 hi) {
    bf16x16 r;
#pragma unroll
    for (int i = 0; i < 8; ++i) { r[i] = lo[i]; r[i + 8] = hi[i]; }
    return r;
}

static __device__ inline bf16x16 cvt16(float4 a, float4 b, float4 c, float4 d) {
    bf16x16 r;
    r[0]  = (__bf16)a.x; r[1]  = (__bf16)a.y; r[2]  = (__bf16)a.z; r[3]  = (__bf16)a.w;
    r[4]  = (__bf16)b.x; r[5]  = (__bf16)b.y; r[6]  = (__bf16)b.z; r[7]  = (__bf16)b.w;
    r[8]  = (__bf16)c.x; r[9]  = (__bf16)c.y; r[10] = (__bf16)c.z; r[11] = (__bf16)c.w;
    r[12] = (__bf16)d.x; r[13] = (__bf16)d.y; r[14] = (__bf16)d.z; r[15] = (__bf16)d.w;
    return r;
}

static __device__ inline unsigned pk2(float a, float b) {
    unsigned short u0 = __builtin_bit_cast(unsigned short, (__bf16)a);
    unsigned short u1 = __builtin_bit_cast(unsigned short, (__bf16)b);
    return (unsigned)u0 | ((unsigned)u1 << 16);
}

#define WMMA_BF16(A, B, C) \
    __builtin_amdgcn_wmma_f32_16x16x32_bf16(false, (A), false, (B), (short)0, (C), false, false)

// ---------------------------------------------------------------------------
// Kernel 1: Y[M,64] = (X[M,1024] @ W[1024,64] + bias) * scale, bf16 output.
// Entire W staged ONCE in LDS (bf16, col-major, padded stride) -> the K loop
// runs with zero barriers: pure global A loads + LDS B loads + WMMA stream.
// 8 waves / 256 threads; each wave: 16 rows x 64 cols, K=1024.
// ---------------------------------------------------------------------------
#define WSTRIDE 1032   // bf16 elements per LDS column row (1024 + 8 pad)
#define WLDS_BYTES (DIMS * WSTRIDE * 2)   // 132096 B (fits 320KB WGP LDS)

__global__ __launch_bounds__(256) void proj_wmma(
    const float* __restrict__ X, const float* __restrict__ W,
    const float* __restrict__ bias, __bf16* __restrict__ Y,
    long ldr, long ldc, float scale)
{
    extern __shared__ __bf16 Wl[];     // [64][WSTRIDE] transposed W, bf16

    const int t = threadIdx.x;
    // --- one-time stage: W[1024][64] fp32 -> Wl[n][k] bf16, packed b32 stores
#pragma unroll 4
    for (int i = 0; i < 32; ++i) {
        const int u  = i * 256 + t;          // 0..8191
        const int n0 = (u & 15) * 4;         // 0..60
        const int k0 = (u >> 4) * 2;         // even k
        float4 a = *(const float4*)(W + (size_t)k0 * DIMS + n0);
        float4 b = *(const float4*)(W + (size_t)(k0 + 1) * DIMS + n0);
        *(unsigned*)&Wl[(n0 + 0) * WSTRIDE + k0] = pk2(a.x, b.x);
        *(unsigned*)&Wl[(n0 + 1) * WSTRIDE + k0] = pk2(a.y, b.y);
        *(unsigned*)&Wl[(n0 + 2) * WSTRIDE + k0] = pk2(a.z, b.z);
        *(unsigned*)&Wl[(n0 + 3) * WSTRIDE + k0] = pk2(a.w, b.w);
    }
    __syncthreads();

    const int lane    = t & 31;
    const int wave    = t >> 5;
    const int hi      = lane >> 4;
    const int l15     = lane & 15;
    const int rowTile = blockIdx.x * 128 + wave * 16;
    const int kkb     = hi ? 8 : 0;

    f32x8 acc[4] = {};
    const float* xrow = X + (size_t)(rowTile + l15) * D_MODEL + (hi ? 8 : 0);

#pragma unroll 2
    for (int k0 = 0; k0 < D_MODEL; k0 += 32) {
        // A fragment (16x32): per-lane K split {0..7,16..23} / {8..15,24..31}
        const float* xp = xrow + k0;
        float4 a0 = *(const float4*)(xp);
        float4 a1 = *(const float4*)(xp + 4);
        float4 a2 = *(const float4*)(xp + 16);
        float4 a3 = *(const float4*)(xp + 20);
        // B fragments: preload all 4 so DS waits drain progressively
        bf16x16 bfr[4];
#pragma unroll
        for (int nt = 0; nt < 4; ++nt) {
            const __bf16* wp = &Wl[(nt * 16 + l15) * WSTRIDE + k0 + kkb];
            bfr[nt] = bjoin(*(const bf16x8*)(wp), *(const bf16x8*)(wp + 16));
        }
        bf16x16 afrag = cvt16(a0, a1, a2, a3);
#pragma unroll
        for (int nt = 0; nt < 4; ++nt)
            acc[nt] = WMMA_BF16(afrag, bfr[nt], acc[nt]);
    }

#pragma unroll
    for (int nt = 0; nt < 4; ++nt) {
        const int col = nt * 16 + l15;
        const float bv = bias[col];
#pragma unroll
        for (int j = 0; j < 8; ++j) {
            const int row = rowTile + j + hi * 8;
            float v = (acc[nt][j] + bv) * scale;
            Y[(size_t)row * ldr + (size_t)col * ldc] = (__bf16)v;
        }
    }
}

// ---------------------------------------------------------------------------
// Kernel 2: flash attention, split-K over Sk (blockIdx.y = partition).
// One wave owns 16 query rows and streams its 1024-key partition in 64-key
// chunks (16 WMMAs per chunk). Writes fp32 partial (acc, m, l) to workspace.
// ---------------------------------------------------------------------------
#define AT_WAVES 4
#define PST 72   // P LDS row stride (bf16): 16B-aligned rows, conflict-free

__global__ __launch_bounds__(AT_WAVES * 32) void flash_attn(
    const __bf16* __restrict__ Q, const __bf16* __restrict__ K,
    const __bf16* __restrict__ VT, float* __restrict__ pacc,
    float* __restrict__ pmv, float* __restrict__ plv)
{
    __shared__ __bf16 Plds[AT_WAVES][16 * PST];

    const int lane  = threadIdx.x & 31;
    const int wave  = threadIdx.x >> 5;
    const int hi    = lane >> 4;
    const int l15   = lane & 15;
    const int part  = blockIdx.y;
    const int qr0   = (blockIdx.x * AT_WAVES + wave) * 16;
    const int b     = qr0 / SQ;
    const int kkb   = hi ? 8 : 0;
    __bf16* pb = Plds[wave];

    // Q A-fragments, head-dim 0..31 / 32..63 (Q pre-scaled by dims^-0.5)
    const __bf16* qrow = Q + (size_t)(qr0 + l15) * DIMS + (hi ? 8 : 0);
    bf16x16 qa0 = bjoin(*(const bf16x8*)(qrow),      *(const bf16x8*)(qrow + 16));
    bf16x16 qa1 = bjoin(*(const bf16x8*)(qrow + 32), *(const bf16x8*)(qrow + 48));

    f32x8 acc[4] = {};
    float mrun[8], lrun[8];
#pragma unroll
    for (int j = 0; j < 8; ++j) { mrun[j] = -1e30f; lrun[j] = 0.0f; }

    const __bf16* kbB   = K + ((size_t)b * SK + (size_t)part * SKP) * DIMS;
    const size_t  vcol0 = (size_t)b * SK + (size_t)part * SKP;

    for (int s0 = 0; s0 < SKP; s0 += CHUNK) {
        // ---- scores: S[16 x 64] = Qtile @ Kchunk^T  (8 WMMAs) ----
        f32x8 sc[4];
#pragma unroll
        for (int st = 0; st < 4; ++st) {
            const __bf16* kr = kbB + (size_t)(s0 + st * 16 + l15) * DIMS + kkb;
            bf16x16 f0 = bjoin(*(const bf16x8*)(kr),      *(const bf16x8*)(kr + 16));
            bf16x16 f1 = bjoin(*(const bf16x8*)(kr + 32), *(const bf16x8*)(kr + 48));
            f32x8 s = {};
            s = WMMA_BF16(qa0, f0, s);
            s = WMMA_BF16(qa1, f1, s);
            sc[st] = s;
        }
        __builtin_prefetch(kbB + (size_t)(s0 + CHUNK + l15) * DIMS, 0, 0);

        // ---- online softmax (rows live in 16-lane halves; xor<=8 stays in half)
        float cm[8];
#pragma unroll
        for (int j = 0; j < 8; ++j)
            cm[j] = fmaxf(fmaxf(sc[0][j], sc[1][j]), fmaxf(sc[2][j], sc[3][j]));
#pragma unroll
        for (int off = 1; off < 16; off <<= 1)
#pragma unroll
            for (int j = 0; j < 8; ++j) cm[j] = fmaxf(cm[j], __shfl_xor(cm[j], off, 32));

        float al[8], rs[8];
#pragma unroll
        for (int j = 0; j < 8; ++j) {
            const float mn = fmaxf(mrun[j], cm[j]);
            al[j] = __expf(mrun[j] - mn);
            float p0 = __expf(sc[0][j] - mn);
            float p1 = __expf(sc[1][j] - mn);
            float p2 = __expf(sc[2][j] - mn);
            float p3 = __expf(sc[3][j] - mn);
            sc[0][j] = p0; sc[1][j] = p1; sc[2][j] = p2; sc[3][j] = p3;
            rs[j] = (p0 + p1) + (p2 + p3);
            mrun[j] = mn;
        }
#pragma unroll
        for (int off = 1; off < 16; off <<= 1)
#pragma unroll
            for (int j = 0; j < 8; ++j) rs[j] += __shfl_xor(rs[j], off, 32);
#pragma unroll
        for (int j = 0; j < 8; ++j) lrun[j] = lrun[j] * al[j] + rs[j];
#pragma unroll
        for (int nt = 0; nt < 4; ++nt)
#pragma unroll
            for (int j = 0; j < 8; ++j) acc[nt][j] *= al[j];

        // ---- P: C-layout -> A-layout via wave-private LDS ----
#pragma unroll
        for (int st = 0; st < 4; ++st)
#pragma unroll
            for (int j = 0; j < 8; ++j) {
                const int r = j + hi * 8;
                pb[r * PST + st * 16 + l15] = (__bf16)sc[st][j];
            }
        const __bf16* prd = pb + l15 * PST + (hi ? 8 : 0);
        bf16x16 pa0 = bjoin(*(const bf16x8*)(prd),      *(const bf16x8*)(prd + 16));
        bf16x16 pa1 = bjoin(*(const bf16x8*)(prd + 32), *(const bf16x8*)(prd + 48));

        // ---- out += P @ Vchunk  (8 WMMAs, contiguous V^T fragments) ----
        const size_t vc = vcol0 + s0 + (hi ? 8 : 0);
#pragma unroll
        for (int nt = 0; nt < 4; ++nt) {
            const __bf16* vp = VT + (size_t)(nt * 16 + l15) * MK + vc;
            bf16x16 v0 = bjoin(*(const bf16x8*)(vp),      *(const bf16x8*)(vp + 16));
            bf16x16 v1 = bjoin(*(const bf16x8*)(vp + 32), *(const bf16x8*)(vp + 48));
            acc[nt] = WMMA_BF16(pa0, v0, acc[nt]);
            acc[nt] = WMMA_BF16(pa1, v1, acc[nt]);
        }
    }

    // ---- store fp32 partials for this partition ----
    const size_t pbase = (size_t)part * MQ;
#pragma unroll
    for (int nt = 0; nt < 4; ++nt) {
        const int col = nt * 16 + l15;
#pragma unroll
        for (int j = 0; j < 8; ++j) {
            const int row = qr0 + j + hi * 8;
            pacc[(pbase + row) * DIMS + col] = acc[nt][j];
        }
    }
    if (l15 == 0) {
#pragma unroll
        for (int j = 0; j < 8; ++j) {
            const int row = qr0 + j + hi * 8;
            pmv[pbase + row] = mrun[j];
            plv[pbase + row] = lrun[j];
        }
    }
}

// ---------------------------------------------------------------------------
// Kernel 3: combine NSPLIT softmax partials -> final fp32 output.
// ---------------------------------------------------------------------------
__global__ __launch_bounds__(256) void attn_combine(
    const float* __restrict__ pacc, const float* __restrict__ pmv,
    const float* __restrict__ plv, float* __restrict__ O)
{
    const int idx = blockIdx.x * 256 + threadIdx.x;   // row*64 + col
    const int row = idx >> 6;
    float m0 = pmv[row], m1 = pmv[MQ + row], m2 = pmv[2 * MQ + row], m3 = pmv[3 * MQ + row];
    float M  = fmaxf(fmaxf(m0, m1), fmaxf(m2, m3));
    float w0 = __expf(m0 - M), w1 = __expf(m1 - M);
    float w2 = __expf(m2 - M), w3 = __expf(m3 - M);
    float den = w0 * plv[row] + w1 * plv[MQ + row] + w2 * plv[2 * MQ + row] + w3 * plv[3 * MQ + row];
    float num = w0 * pacc[idx]
              + w1 * pacc[(size_t)MQ * DIMS + idx]
              + w2 * pacc[(size_t)2 * MQ * DIMS + idx]
              + w3 * pacc[(size_t)3 * MQ * DIMS + idx];
    O[idx] = num / den;
}

// ---------------------------------------------------------------------------
extern "C" void kernel_launch(void* const* d_in, const int* in_sizes, int n_in,
                              void* d_out, int out_size, void* d_ws, size_t ws_size,
                              hipStream_t stream) {
    const float* enc = (const float*)d_in[0];
    const float* dec = (const float*)d_in[1];
    const float* Wq  = (const float*)d_in[2];
    const float* bq  = (const float*)d_in[3];
    const float* Wk  = (const float*)d_in[4];
    const float* bk  = (const float*)d_in[5];
    const float* Wv  = (const float*)d_in[6];
    const float* bv  = (const float*)d_in[7];
    float* out = (float*)d_out;

    // workspace layout (13.3 MB total)
    __bf16* qb   = (__bf16*)d_ws;                       // [MQ,64]      1 MB
    __bf16* kbm  = qb  + (size_t)MQ * DIMS;             // [MK,64]      2 MB
    __bf16* vtb  = kbm + (size_t)MK * DIMS;             // [64,MK]      2 MB (transposed)
    float*  pacc = (float*)(vtb + (size_t)MK * DIMS);   // [NSPLIT,MQ,64]  8 MB
    float*  pmv  = pacc + (size_t)NSPLIT * MQ * DIMS;   // [NSPLIT,MQ]   128 KB
    float*  plv  = pmv  + (size_t)NSPLIT * MQ;          // [NSPLIT,MQ]   128 KB

    const float qscale = 0.125f;   // DIMS^-0.5 folded into Q projection

    proj_wmma<<<MQ / 128, 256, WLDS_BYTES, stream>>>(dec, Wq, bq, qb,  DIMS, 1, qscale);
    proj_wmma<<<MK / 128, 256, WLDS_BYTES, stream>>>(enc, Wk, bk, kbm, DIMS, 1, 1.0f);
    proj_wmma<<<MK / 128, 256, WLDS_BYTES, stream>>>(enc, Wv, bv, vtb, 1, MK,  1.0f);

    dim3 fgrid(MQ / (16 * AT_WAVES), NSPLIT);
    flash_attn<<<fgrid, AT_WAVES * 32, 0, stream>>>(qb, kbm, vtb, pacc, pmv, plv);

    attn_combine<<<(MQ * DIMS) / 256, 256, 0, stream>>>(pacc, pmv, plv, out);
}
